// SelectiveScan2D_16312285791111
// MI455X (gfx1250) — matrix operand order repaired and checked by reading, not compiled, not run on hardware
//
#include <hip/hip_runtime.h>
#include <hip/hip_bf16.h>

typedef __attribute__((ext_vector_type(16))) _Float16 v16h;
typedef __attribute__((ext_vector_type(8)))  float    v8f;

#define BSZ 8
#define CC  256
#define NN  64
#define LL  4096           // H*W
#define NTOK (BSZ*LL)      // 32768

// ---------------------------------------------------------------------------
// Kernel 1: double layer-norm per token (ref applies LN once because L>4000,
// then again inside each scan). One wave per token, 8 channels per lane.
// ---------------------------------------------------------------------------
__global__ __launch_bounds__(256) void ln_kernel(const float* __restrict__ x,
                                                 _Float16* __restrict__ xn) {
  const int wave = threadIdx.x >> 5;
  const int lane = threadIdx.x & 31;
  const size_t tok = (size_t)blockIdx.x * 8 + wave;
  const float* xp = x + tok * CC;

  float v[8], s = 0.f, s2 = 0.f;
#pragma unroll
  for (int j = 0; j < 8; ++j) { float t = xp[lane + 32 * j]; v[j] = t; s += t; s2 += t * t; }
#pragma unroll
  for (int off = 16; off > 0; off >>= 1) { s += __shfl_xor(s, off, 32); s2 += __shfl_xor(s2, off, 32); }
  float m   = s * (1.f / CC);
  float var = fmaxf(s2 * (1.f / CC) - m * m, 0.f);
  float inv = rsqrtf(var + 1e-5f);

  s = 0.f; s2 = 0.f;
#pragma unroll
  for (int j = 0; j < 8; ++j) { float t = (v[j] - m) * inv; v[j] = t; s += t; s2 += t * t; }
#pragma unroll
  for (int off = 16; off > 0; off >>= 1) { s += __shfl_xor(s, off, 32); s2 += __shfl_xor(s2, off, 32); }
  float m2   = s * (1.f / CC);
  float var2 = fmaxf(s2 * (1.f / CC) - m2 * m2, 0.f);
  float inv2 = rsqrtf(var2 + 1e-5f);

  _Float16* op = xn + tok * CC;
#pragma unroll
  for (int j = 0; j < 8; ++j) op[lane + 32 * j] = (_Float16)((v[j] - m2) * inv2);
}

// ---------------------------------------------------------------------------
// Kernel 2: u = xn @ W_in^T  via WMMA f32_16x16x32_f16.
// One wave computes a 16-token x 16-n tile, K loop over C=256 (8 steps).
// ---------------------------------------------------------------------------
__global__ __launch_bounds__(32) void uproj_kernel(const _Float16* __restrict__ xn,
                                                   const float* __restrict__ Win,
                                                   float* __restrict__ u) {
  const int tile = blockIdx.x;          // 16-token tile (2048 tiles)
  const int nt   = blockIdx.y;          // 16-col tile of N (4 tiles)
  const int lane = threadIdx.x;
  const int row  = lane & 15;
  const int half = lane >> 4;

  v8f acc = {0.f, 0.f, 0.f, 0.f, 0.f, 0.f, 0.f, 0.f};
  const _Float16* abase = xn + ((size_t)tile * 16 + row) * CC + half * 8;
  const float*    bbase = Win + ((size_t)nt * 16 + row) * CC + half * 8;

#pragma unroll
  for (int ks = 0; ks < 8; ++ks) {
    v16h a, b;
    const _Float16* ap = abase + ks * 32;
#pragma unroll
    for (int j = 0; j < 8; ++j) { a[j] = ap[j]; a[8 + j] = ap[16 + j]; }
    const float* bp = bbase + ks * 32;
#pragma unroll
    for (int j = 0; j < 8; ++j) { b[j] = (_Float16)bp[j]; b[8 + j] = (_Float16)bp[16 + j]; }
    acc = __builtin_amdgcn_wmma_f32_16x16x32_f16(false, a, false, b, (short)0, acc,
                                                 false, false);
  }
#pragma unroll
  for (int r = 0; r < 8; ++r) {
    int M = r + half * 8;
    u[((size_t)tile * 16 + M) * NN + nt * 16 + (lane & 15)] = acc[r];
  }
}

// ---------------------------------------------------------------------------
// Scan machinery: h_t = Ad*h_{t-1} + Bp*u_t over 4 token orderings.
// dir0 = row fwd, dir1 = row bwd, dir2 = col fwd, dir3 = col bwd.
// ---------------------------------------------------------------------------
__device__ __forceinline__ float compute_Ad(float Aval, float snr) {
  float s     = fminf(fmaxf(snr, 0.f), 1.f);
  float scale = 0.6f + 0.4f * s;
  float ad    = tanhf(Aval) * scale;
  return fminf(fmaxf(ad, -0.99f), 0.99f);
}

__device__ __forceinline__ int tok_of(int dir, int c, int i) {
  switch (dir) {
    case 0:  return c * 64 + i;
    case 1:  return 4095 - (c * 64 + i);
    case 2:  return i * 64 + c;
    default: return (63 - i) * 64 + (63 - c);
  }
}

// Pass 1: per-chunk local scans (h0 = 0); 64 chunks of 64 steps per sequence.
__global__ __launch_bounds__(64) void scan_pass1(const float* __restrict__ u,
                                                 const float* __restrict__ A,
                                                 const float* __restrict__ snr,
                                                 const float* __restrict__ Bp,
                                                 float* __restrict__ local) {
  const int idx = blockIdx.x;   // ((dir*8)+b)*64 + c
  const int c   = idx & 63;
  const int db  = idx >> 6;
  const int b   = db & 7;
  const int dir = db >> 3;
  const int n   = threadIdx.x;

  const float ad = compute_Ad(A[n], snr[b]);
  const float bp = Bp[n];
  const float* ub = u + (size_t)b * LL * NN + n;

  float h = 0.f;
  for (int i = 0; i < 64; ++i)
    h = fmaf(ad, h, bp * ub[(size_t)tok_of(dir, c, i) * NN]);
  local[(size_t)idx * 64 + n] = h;
}

// Pass 2: carry propagation across chunks using P = Ad^64 (6 squarings).
__global__ __launch_bounds__(64) void scan_pass2(const float* __restrict__ local,
                                                 const float* __restrict__ A,
                                                 const float* __restrict__ snr,
                                                 float* __restrict__ carry) {
  const int idx = blockIdx.x * 64 + threadIdx.x;  // (dir*8+b)*64 + n
  const int n   = idx & 63;
  const int db  = idx >> 6;
  const int b   = db & 7;

  const float ad = compute_Ad(A[n], snr[b]);
  float P = ad;
#pragma unroll
  for (int i = 0; i < 6; ++i) P = P * P;          // Ad^64

  float h = 0.f;
  for (int c = 0; c < 64; ++c) {
    size_t k = ((size_t)db * 64 + c) * 64 + n;
    carry[k] = h;                                  // state entering chunk c
    h = fmaf(P, h, local[k]);
  }
}

// Pass 3: replay fwd+bwd of one orientation over a 64-token chunk, accumulate
// Cp*h contributions in LDS (per-thread column, race-free), write 0.25*sum.
__global__ __launch_bounds__(64) void scan_pass3(const float* __restrict__ u,
                                                 const float* __restrict__ carry,
                                                 const float* __restrict__ A,
                                                 const float* __restrict__ snr,
                                                 const float* __restrict__ Bp,
                                                 const float* __restrict__ Cp,
                                                 float* __restrict__ sout,
                                                 int mode /*0=row,1=col*/) {
  __shared__ float tile[64 * 65];
  const int c = blockIdx.x & 63;
  const int b = blockIdx.x >> 6;
  const int n = threadIdx.x;

  const float ad = compute_Ad(A[n], snr[b]);
  const float bp = Bp[n];
  const float cp = Cp[n];
  const int dirF = mode ? 2 : 0;
  const float* ub = u + (size_t)b * LL * NN + n;

  // forward direction: token chunk c == scan chunk c
  float h = carry[(((size_t)dirF * 8 + b) * 64 + c) * 64 + n];
  for (int i = 0; i < 64; ++i) {
    int tok = mode ? (i * 64 + c) : (c * 64 + i);
    h = fmaf(ad, h, bp * ub[(size_t)tok * NN]);
    tile[i * 65 + n] = cp * h;
  }
  // backward direction: token chunk c == scan chunk 63-c; iterate tokens descending
  h = carry[(((size_t)(dirF + 1) * 8 + b) * 64 + (63 - c)) * 64 + n];
  for (int i = 63; i >= 0; --i) {
    int tok = mode ? (i * 64 + c) : (c * 64 + i);
    h = fmaf(ad, h, bp * ub[(size_t)tok * NN]);
    tile[i * 65 + n] += cp * h;
  }
  for (int i = 0; i < 64; ++i) {
    int tok = mode ? (i * 64 + c) : (c * 64 + i);
    sout[((size_t)b * LL + tok) * NN + n] = 0.25f * tile[i * 65 + n];
  }
}

// ---------------------------------------------------------------------------
// Kernel 6: y = (s_row + s_col) @ W_out^T + D*xn  via WMMA (K = N = 64).
// ---------------------------------------------------------------------------
__global__ __launch_bounds__(32) void outproj_kernel(const float* __restrict__ sr,
                                                     const float* __restrict__ sc,
                                                     const float* __restrict__ Wout,
                                                     const float* __restrict__ D,
                                                     const _Float16* __restrict__ xn,
                                                     float* __restrict__ out) {
  const int tile = blockIdx.x;          // 16-token tile
  const int ct   = blockIdx.y;          // 16-channel tile of C (16 tiles)
  const int lane = threadIdx.x;
  const int row  = lane & 15;
  const int half = lane >> 4;

  v8f acc = {0.f, 0.f, 0.f, 0.f, 0.f, 0.f, 0.f, 0.f};
#pragma unroll
  for (int ks = 0; ks < 2; ++ks) {
    v16h a, bvec;
    size_t ab = ((size_t)tile * 16 + row) * NN + ks * 32 + half * 8;
#pragma unroll
    for (int j = 0; j < 8; ++j) {
      a[j]     = (_Float16)(sr[ab + j]      + sc[ab + j]);
      a[8 + j] = (_Float16)(sr[ab + 16 + j] + sc[ab + 16 + j]);
    }
    size_t bb = ((size_t)ct * 16 + row) * NN + ks * 32 + half * 8;  // W_out[c][k]
#pragma unroll
    for (int j = 0; j < 8; ++j) { bvec[j] = (_Float16)Wout[bb + j]; bvec[8 + j] = (_Float16)Wout[bb + 16 + j]; }
    acc = __builtin_amdgcn_wmma_f32_16x16x32_f16(false, a, false, bvec, (short)0, acc,
                                                 false, false);
  }
  const int col = ct * 16 + (lane & 15);
  const float d = D[col];
#pragma unroll
  for (int r = 0; r < 8; ++r) {
    int M = r + half * 8;
    size_t gt = (size_t)tile * 16 + M;
    out[gt * CC + col] = acc[r] + d * (float)xn[gt * CC + col];
  }
}

// ---------------------------------------------------------------------------
extern "C" void kernel_launch(void* const* d_in, const int* in_sizes, int n_in,
                              void* d_out, int out_size, void* d_ws, size_t ws_size,
                              hipStream_t stream) {
  (void)in_sizes; (void)n_in; (void)out_size; (void)ws_size;
  const float* x    = (const float*)d_in[0];
  const float* snr  = (const float*)d_in[1];
  const float* A    = (const float*)d_in[2];
  const float* Bp   = (const float*)d_in[3];
  const float* Cp   = (const float*)d_in[4];
  const float* D    = (const float*)d_in[5];
  const float* Win  = (const float*)d_in[6];
  const float* Wout = (const float*)d_in[7];
  float* out = (float*)d_out;

  char* ws = (char*)d_ws;
  _Float16* xn  = (_Float16*)(ws);                                   // 16 MiB
  float* u      = (float*)(ws + (size_t)16 * 1024 * 1024);           //  8 MiB
  float* s_row  = (float*)(ws + (size_t)24 * 1024 * 1024);           //  8 MiB
  float* s_col  = (float*)(ws + (size_t)32 * 1024 * 1024);           //  8 MiB
  float* local  = (float*)(ws + (size_t)40 * 1024 * 1024);           // 512 KiB
  float* carry  = (float*)(ws + (size_t)40 * 1024 * 1024 + 512 * 1024);

  ln_kernel<<<NTOK / 8, 256, 0, stream>>>(x, xn);
  uproj_kernel<<<dim3(NTOK / 16, NN / 16), 32, 0, stream>>>(xn, Win, u);
  scan_pass1<<<4 * BSZ * 64, 64, 0, stream>>>(u, A, snr, Bp, local);
  scan_pass2<<<(4 * BSZ * 64) / 64, 64, 0, stream>>>(local, A, snr, carry);
  scan_pass3<<<BSZ * 64, 64, 0, stream>>>(u, carry, A, snr, Bp, Cp, s_row, 0);
  scan_pass3<<<BSZ * 64, 64, 0, stream>>>(u, carry, A, snr, Bp, Cp, s_col, 1);
  outproj_kernel<<<dim3(NTOK / 16, CC / 16), 32, 0, stream>>>(s_row, s_col, Wout, D, xn, out);
}
